// Discriminator_10213432229968
// MI455X (gfx1250) — matrix operand order repaired and checked by reading, not compiled, not run on hardware
//
#include <hip/hip_runtime.h>
#include <math.h>

#define N_NODES  100000
#define N_EDGES  3200000
#define N_GRAPHS 64
#define F1 64
#define F2 32
#define ROW_TILES (N_NODES / 16)   // 6250 exact

typedef __attribute__((ext_vector_type(2)))  float    v2f;
typedef __attribute__((ext_vector_type(8)))  float    v8f;
typedef __attribute__((ext_vector_type(16))) _Float16 v16h;

// ---------------------------------------------------------------------------
// 1) init: deg = 1 (self loop), pooling accumulators = 0
__global__ void k_init(float* __restrict__ deg, float* __restrict__ pool_sum,
                       float* __restrict__ pool_cnt) {
  int i = blockIdx.x * blockDim.x + threadIdx.x;
  if (i < N_NODES) deg[i] = 1.0f;
  if (i < N_GRAPHS * F2) pool_sum[i] = 0.0f;
  if (i < N_GRAPHS) pool_cnt[i] = 0.0f;
}

// 2) degree of dst side (symmetric-norm GCN uses in-degree of dst2)
__global__ void k_deg(const int* __restrict__ dst, float* __restrict__ deg) {
  int e = blockIdx.x * blockDim.x + threadIdx.x;
  if (e < N_EDGES) atomicAdd(&deg[dst[e]], 1.0f);
}

// 3) dinv = rsqrt(deg) (deg >= 1 always); xs = x*dinv; agg1 seeded with self loop
__global__ void k_node1(const float* __restrict__ x, const float* __restrict__ deg,
                        float* __restrict__ dinv, float* __restrict__ xs,
                        float* __restrict__ agg1) {
  int n = blockIdx.x * blockDim.x + threadIdx.x;
  if (n < N_NODES) {
    float di = rsqrtf(deg[n]);
    dinv[n] = di;
    float v = x[n] * di;
    xs[n] = v;
    agg1[n] = v;            // self-loop contribution x[n]*dinv[n]
  }
}

// 4) layer-1 aggregation is SCALAR per edge (1 input feature factors out)
__global__ void k_agg1(const int* __restrict__ src, const int* __restrict__ dst,
                       const float* __restrict__ xs, float* __restrict__ agg1) {
  int e = blockIdx.x * blockDim.x + threadIdx.x;
  if (e < N_EDGES) atomicAdd(&agg1[dst[e]], xs[src[e]]);
}

// 5) H1[n][j] = relu(W1[j]*dinv[n]*agg1[n] + b1[j])
__global__ void k_h1(const float* __restrict__ agg1, const float* __restrict__ dinv,
                     const float* __restrict__ W1, const float* __restrict__ b1,
                     float* __restrict__ H1) {
  int i = blockIdx.x * blockDim.x + threadIdx.x;   // n*64 + j
  int n = i >> 6, j = i & 63;
  if (n < N_NODES) {
    float v = W1[j] * dinv[n] * agg1[n] + b1[j];
    H1[i] = fmaxf(v, 0.0f);
  }
}

// ---------------------------------------------------------------------------
// 6) GEMM  G = (H1 @ W2) * dinv[n]  via V_WMMA_F32_16X16X4_F32.
//    One wave per 16-row tile, two 16x16 N-tiles (F2=32), K=64 in 16 steps.
//    Epilogue also seeds scatter accumulator S with the self-loop term.
__global__ void __launch_bounds__(128) k_gemm(const float* __restrict__ H1,
                                              const float* __restrict__ W2,
                                              const float* __restrict__ dinv,
                                              float* __restrict__ G,
                                              float* __restrict__ S) {
  __shared__ float Ws[F1 * F2];                    // 8 KB of LDS for W2
  for (int i = threadIdx.x; i < F1 * F2; i += blockDim.x) Ws[i] = W2[i];
  __syncthreads();

  int wave = blockIdx.x * (blockDim.x >> 5) + (threadIdx.x >> 5);
  if (wave >= ROW_TILES) return;                   // wave-uniform: EXEC stays all-1s
  int lane = threadIdx.x & 31;
  int lr   = lane & 15;                            // row (A) / col (B,D) within half
  int hi   = lane >> 4;                            // upper half-wave flag
  int row0 = wave * 16;
  const float* Arow = H1 + (size_t)(row0 + lr) * F1;

  v8f acc0 = {0.f,0.f,0.f,0.f,0.f,0.f,0.f,0.f};
  v8f acc1 = {0.f,0.f,0.f,0.f,0.f,0.f,0.f,0.f};

#if __has_builtin(__builtin_amdgcn_wmma_f32_16x16x4_f32)
  // A 16x4 f32: lanes 0-15 hold K=k0,k0+1 ; lanes 16-31 hold K=k0+2,k0+3
  // B 4x16 f32: lanes 0-15 col N=lr with K=k0,k0+1 ; lanes 16-31 K=k0+2,k0+3
  for (int k0 = 0; k0 < F1; k0 += 4) {
    int ka = k0 + hi * 2;
    v2f a;  a.x = Arow[ka];              a.y = Arow[ka + 1];
    v2f b0; b0.x = Ws[ka * F2 + lr];     b0.y = Ws[(ka + 1) * F2 + lr];
    v2f b1; b1.x = Ws[ka * F2 + 16 + lr];b1.y = Ws[(ka + 1) * F2 + 16 + lr];
    acc0 = __builtin_amdgcn_wmma_f32_16x16x4_f32(false, a, false, b0, (short)0,
                                                 acc0, false, false);
    acc1 = __builtin_amdgcn_wmma_f32_16x16x4_f32(false, a, false, b1, (short)0,
                                                 acc1, false, false);
  }
#else
  // Fallback: probe-confirmed f16 WMMA (f32 accumulate), two K=32 steps
  for (int kb = 0; kb < F1; kb += 32) {
    v16h af, bf0, bf1;
    for (int p = 0; p < 8; ++p) {
      af[p]     = (_Float16)Arow[kb + hi * 8 + p];
      af[p + 8] = (_Float16)Arow[kb + 16 + hi * 8 + p];
    }
    for (int p = 0; p < 16; ++p) {
      int kk = kb + (p < 8 ? hi * 8 + p : 16 + hi * 8 + (p - 8));
      bf0[p] = (_Float16)Ws[kk * F2 + lr];
      bf1[p] = (_Float16)Ws[kk * F2 + 16 + lr];
    }
    acc0 = __builtin_amdgcn_wmma_f32_16x16x32_f16(false, af, false, bf0, (short)0,
                                                  acc0, false, false);
    acc1 = __builtin_amdgcn_wmma_f32_16x16x32_f16(false, af, false, bf1, (short)0,
                                                  acc1, false, false);
  }
#endif

  // D layout: VGPR r -> M = r + hi*8, N = lr.  Fold src-side norm dinv[node].
  for (int r = 0; r < 8; ++r) {
    int node = row0 + r + hi * 8;
    float di = dinv[node];
    float g0 = acc0[r] * di;
    float g1 = acc1[r] * di;
    size_t o = (size_t)node * F2;
    G[o + lr] = g0;       S[o + lr] = g0;       // S seeded with self-loop term
    G[o + 16 + lr] = g1;  S[o + 16 + lr] = g1;
  }
}

// ---------------------------------------------------------------------------
// 7) layer-2 edge scatter: one wave per edge, lane = feature (32 = warpSize)
__global__ void k_agg2(const int* __restrict__ src, const int* __restrict__ dst,
                       const float* __restrict__ G, float* __restrict__ S) {
  int gid = blockIdx.x * blockDim.x + threadIdx.x;
  int e = gid >> 5, lane = gid & 31;
  if (e < N_EDGES) {
    int s = src[e], d = dst[e];
    float v = G[(size_t)s * F2 + lane];          // coalesced 128B gather (L2-resident)
    atomicAdd(&S[(size_t)d * F2 + lane], v);     // global_atomic_add_f32
  }
}

// 8) bias+relu and mean-pool accumulation: one wave per node, lane = feature
__global__ void k_pool(const float* __restrict__ S, const float* __restrict__ dinv,
                       const float* __restrict__ b2, const int* __restrict__ batch,
                       float* __restrict__ pool_sum, float* __restrict__ pool_cnt) {
  int gid = blockIdx.x * blockDim.x + threadIdx.x;
  int n = gid >> 5, lane = gid & 31;
  if (n < N_NODES) {
    float v = fmaxf(dinv[n] * S[(size_t)n * F2 + lane] + b2[lane], 0.0f);
    int g = batch[n];
    atomicAdd(&pool_sum[g * F2 + lane], v);
    if (lane == 0) atomicAdd(&pool_cnt[g], 1.0f);
  }
}

// 9) FC head: mean, dot with Wfc via wave32 shuffle reduction, sigmoid
__global__ void k_head(const float* __restrict__ pool_sum,
                       const float* __restrict__ pool_cnt,
                       const float* __restrict__ Wfc, const float* __restrict__ bfc,
                       float* __restrict__ out) {
  int w = threadIdx.x >> 5, lane = threadIdx.x & 31;
  for (int g = w; g < N_GRAPHS; g += (int)(blockDim.x >> 5)) {
    float cnt = fmaxf(pool_cnt[g], 1.0f);
    float p = (pool_sum[g * F2 + lane] / cnt) * Wfc[lane];
    for (int off = 16; off > 0; off >>= 1) p += __shfl_xor(p, off, 32);
    if (lane == 0) out[g] = 1.0f / (1.0f + __expf(-(p + bfc[0])));
  }
}

// ---------------------------------------------------------------------------
extern "C" void kernel_launch(void* const* d_in, const int* in_sizes, int n_in,
                              void* d_out, int out_size, void* d_ws, size_t ws_size,
                              hipStream_t stream) {
  const float* x    = (const float*)d_in[0];
  const int*   edge = (const int*)  d_in[1];     // [2, E] row-major
  const int*   batch= (const int*)  d_in[2];
  const float* W1   = (const float*)d_in[3];
  const float* b1   = (const float*)d_in[4];
  const float* W2   = (const float*)d_in[5];
  const float* b2   = (const float*)d_in[6];
  const float* Wfc  = (const float*)d_in[7];
  const float* bfc  = (const float*)d_in[8];
  float* out = (float*)d_out;

  const int* src = edge;
  const int* dst = edge + N_EDGES;

  // workspace layout (floats): ~53 MB total
  float* deg      = (float*)d_ws;                       // N
  float* dinv     = deg      + N_NODES;                 // N
  float* xs       = dinv     + N_NODES;                 // N
  float* agg1     = xs       + N_NODES;                 // N
  float* H1       = agg1     + N_NODES;                 // N*64
  float* G        = H1       + (size_t)N_NODES * F1;    // N*32
  float* S        = G        + (size_t)N_NODES * F2;    // N*32
  float* pool_sum = S        + (size_t)N_NODES * F2;    // 64*32
  float* pool_cnt = pool_sum + N_GRAPHS * F2;           // 64

  const int T = 256;
  k_init <<<(N_NODES + T - 1) / T, T, 0, stream>>>(deg, pool_sum, pool_cnt);
  k_deg  <<<(N_EDGES + T - 1) / T, T, 0, stream>>>(dst, deg);
  k_node1<<<(N_NODES + T - 1) / T, T, 0, stream>>>(x, deg, dinv, xs, agg1);
  k_agg1 <<<(N_EDGES + T - 1) / T, T, 0, stream>>>(src, dst, xs, agg1);
  k_h1   <<<(N_NODES * F1 + T - 1) / T, T, 0, stream>>>(agg1, dinv, W1, b1, H1);
  k_gemm <<<(ROW_TILES + 3) / 4, 128, 0, stream>>>(H1, W2, dinv, G, S);
  {
    long long thr = (long long)N_EDGES * 32;
    k_agg2<<<(unsigned)((thr + T - 1) / T), T, 0, stream>>>(src, dst, G, S);
  }
  k_pool <<<(N_NODES * 32 + T - 1) / T, T, 0, stream>>>(S, dinv, b2, batch,
                                                        pool_sum, pool_cnt);
  k_head <<<1, 256, 0, stream>>>(pool_sum, pool_cnt, Wfc, bfc, out);
}
